// SparseTransformerBlock_68951404970493
// MI455X (gfx1250) — compile-verified, matrix operand order checked by reading, not run on hardware
//
#include <hip/hip_runtime.h>

typedef __bf16 bf16;
typedef __attribute__((ext_vector_type(16))) __bf16 v16bf;
typedef __attribute__((ext_vector_type(8)))  float  v8f;

#define QK_SCALE (0.125f * 1.4426950408889634f)   // 1/sqrt(64) * log2(e), folded into Q

// ---------------------------------------------------------------------------
// CDNA5 async global->LDS copy (ASYNCcnt-tracked), with sync fallback.
// Builtin signature (clang gfx1250): void(v4i AS1*, v4i AS3*, Imm int, Imm int)
// ---------------------------------------------------------------------------
#if defined(__HIP_DEVICE_COMPILE__) && \
    __has_builtin(__builtin_amdgcn_global_load_async_to_lds_b128) && \
    __has_builtin(__builtin_amdgcn_s_wait_asynccnt)
#define HAVE_ASYNC_LDS 1
#else
#define HAVE_ASYNC_LDS 0
#endif

typedef int v4i_g __attribute__((vector_size(16)));
typedef __attribute__((address_space(1))) v4i_g* as1_v4i;
typedef __attribute__((address_space(3))) v4i_g* as3_v4i;

__device__ __forceinline__ void async_cp16(const void* g, void* l)
{
#if HAVE_ASYNC_LDS
    __builtin_amdgcn_global_load_async_to_lds_b128((as1_v4i)(g), (as3_v4i)(l), 0, 0);
#else
    *(uint4*)l = *(const uint4*)g;
#endif
}

__device__ __forceinline__ void async_wait0()
{
#if HAVE_ASYNC_LDS
    __builtin_amdgcn_s_wait_asynccnt(0);
#endif
}

// ---------------------------------------------------------------------------
// Weight transpose + fp32 -> bf16 convert:  W[K][N] (row-major) -> Wt[N][K]
// ---------------------------------------------------------------------------
__global__ __launch_bounds__(256)
void transpose_to_bf16(const float* __restrict__ W, bf16* __restrict__ Wt,
                       int K, int N)
{
    __shared__ float t[32][33];
    const int tx = threadIdx.x & 31;
    const int ty = threadIdx.x >> 5;            // 0..7
    const int n0 = blockIdx.x * 32;
    const int k0 = blockIdx.y * 32;
#pragma unroll
    for (int i = 0; i < 32; i += 8)
        t[ty + i][tx] = W[(size_t)(k0 + ty + i) * N + n0 + tx];
    __syncthreads();
#pragma unroll
    for (int i = 0; i < 32; i += 8)
        Wt[(size_t)(n0 + ty + i) * K + k0 + tx] = (bf16)t[tx][ty + i];
}

// ---------------------------------------------------------------------------
// LayerNorm over last dim (1024), fp32 in -> bf16 out.  One block per row.
// ---------------------------------------------------------------------------
__global__ __launch_bounds__(256)
void layernorm_bf16(const float* __restrict__ X, bf16* __restrict__ H)
{
    __shared__ float rs[8], rs2[8];
    const int row = blockIdx.x;
    const int tid = threadIdx.x;
    const float4 v = *(const float4*)&X[(size_t)row * 1024 + tid * 4];

    float s  = v.x + v.y + v.z + v.w;
    float s2 = v.x * v.x + v.y * v.y + v.z * v.z + v.w * v.w;
#pragma unroll
    for (int m = 1; m < 32; m <<= 1) {
        s  += __shfl_xor(s,  m, 32);
        s2 += __shfl_xor(s2, m, 32);
    }
    if ((tid & 31) == 0) { rs[tid >> 5] = s; rs2[tid >> 5] = s2; }
    __syncthreads();
    float ts = 0.f, ts2 = 0.f;
#pragma unroll
    for (int i = 0; i < 8; ++i) { ts += rs[i]; ts2 += rs2[i]; }
    const float mean = ts * (1.f / 1024.f);
    const float var  = ts2 * (1.f / 1024.f) - mean * mean;
    const float inv  = rsqrtf(var + 1e-6f);

    union { bf16 e[4]; uint2 u; } o;
    o.e[0] = (bf16)((v.x - mean) * inv);
    o.e[1] = (bf16)((v.y - mean) * inv);
    o.e[2] = (bf16)((v.z - mean) * inv);
    o.e[3] = (bf16)((v.w - mean) * inv);
    *(uint2*)&H[(size_t)row * 1024 + tid * 4] = o.u;
}

// ---------------------------------------------------------------------------
// Generic bf16 WMMA GEMM:  out = A[M,K] @ Wt[N,K]^T + bias, templated epilogue
//   EP_QKV  : scatter Q (pre-scaled) / K as [B*Hn,L,D], V as [B*Hn,D,L]
//   EP_RESID: out_f32 = acc + bias + resid_f32
//   EP_GELU : out_bf16 = gelu(acc + bias)
// 256 threads (8 waves), tile 128x128, BK=32, double-buffered async staging.
// Wave (wm in 0..1, wn in 0..3) owns a 64x32 region = 4x2 16x16 accums.
// ---------------------------------------------------------------------------
#define EP_QKV   0
#define EP_RESID 1
#define EP_GELU  2

#define LDT 40   // LDS tile row stride in bf16 (80B: 16B-aligned, conflict-free)

template <int EP>
__global__ __launch_bounds__(256)
void gemm_bf16(const bf16* __restrict__ A, const bf16* __restrict__ Bt,
               const float* __restrict__ bias, const float* __restrict__ resid,
               float* __restrict__ outF, bf16* __restrict__ outH,
               bf16* __restrict__ Qo, bf16* __restrict__ Ko, bf16* __restrict__ Vo,
               int M, int N, int K)
{
    __shared__ bf16 As[2][128 * LDT];
    __shared__ bf16 Bs[2][128 * LDT];

    const int tid  = threadIdx.x;
    const int lane = tid & 31;
    const int wave = tid >> 5;
    const int wm   = wave & 1;          // 2 waves along M
    const int wn   = wave >> 1;         // 4 waves along N
    const int half = lane >> 4;
    const int l16  = lane & 15;
    const int m0   = blockIdx.y * 128;
    const int n0   = blockIdx.x * 128;
    const int nk   = K >> 5;

    auto stage = [&](int buf, int k0) {
#pragma unroll
        for (int t = 0; t < 2; ++t) {
            const int ch = tid + t * 256;             // 0..511 : 128 rows x 4 chunks
            const int r  = ch >> 2;
            const int c  = (ch & 3) * 8;
            async_cp16(&A [(size_t)(m0 + r) * K + k0 + c], &As[buf][r * LDT + c]);
            async_cp16(&Bt[(size_t)(n0 + r) * K + k0 + c], &Bs[buf][r * LDT + c]);
        }
    };

    v8f acc[4][2];
#pragma unroll
    for (int i = 0; i < 4; ++i)
#pragma unroll
        for (int j = 0; j < 2; ++j)
#pragma unroll
            for (int r = 0; r < 8; ++r) acc[i][j][r] = 0.f;

    stage(0, 0);
    async_wait0();
    __syncthreads();

    for (int kt = 0; kt < nk; ++kt) {
        const int cur = kt & 1;
        if (kt + 1 < nk) stage(cur ^ 1, (kt + 1) << 5);   // overlap with WMMAs

        v16bf bfrag[2];
#pragma unroll
        for (int j = 0; j < 2; ++j) {
            const int off = (wn * 32 + j * 16 + l16) * LDT + half * 8;
            ((uint4*)&bfrag[j])[0] = *(const uint4*)&Bs[cur][off];
            ((uint4*)&bfrag[j])[1] = *(const uint4*)&Bs[cur][off + 16];
        }
#pragma unroll
        for (int i = 0; i < 4; ++i) {
            const int off = (wm * 64 + i * 16 + l16) * LDT + half * 8;
            v16bf afrag;
            ((uint4*)&afrag)[0] = *(const uint4*)&As[cur][off];
            ((uint4*)&afrag)[1] = *(const uint4*)&As[cur][off + 16];
#pragma unroll
            for (int j = 0; j < 2; ++j)
                acc[i][j] = __builtin_amdgcn_wmma_f32_16x16x32_bf16(
                    false, afrag, false, bfrag[j], (short)0, acc[i][j], false, false);
        }

        async_wait0();          // next tile landed (this wave's issues)
        __syncthreads();        // everyone done reading cur, next buffer visible
    }

    // Epilogue.  Frag element (r, lane): row = r + half*8, col = l16.
#pragma unroll
    for (int i = 0; i < 4; ++i) {
#pragma unroll
        for (int j = 0; j < 2; ++j) {
            const int gn = n0 + wn * 32 + j * 16 + l16;
            const float b = bias[gn];
#pragma unroll
            for (int r = 0; r < 8; ++r) {
                const int gm = m0 + wm * 64 + i * 16 + half * 8 + r;
                float v = acc[i][j][r] + b;
                if (EP == EP_RESID) {
                    outF[(size_t)gm * N + gn] = v + resid[(size_t)gm * N + gn];
                } else if (EP == EP_GELU) {
                    const float u = 0.7978845608028654f * (v + 0.044715f * v * v * v);
                    outH[(size_t)gm * N + gn] = (bf16)(0.5f * v * (1.f + tanhf(u)));
                } else { // EP_QKV: N = 3072; B=4, Hn=16, L=2048, D=64
                    const int which = gn >> 10;
                    const int cc    = gn & 1023;
                    const int h     = cc >> 6;
                    const int d     = cc & 63;
                    const int bb    = gm >> 11;
                    const int l     = gm & 2047;
                    const int bh    = bb * 16 + h;
                    if (which == 0)
                        Qo[((size_t)bh * 2048 + l) * 64 + d] = (bf16)(v * QK_SCALE);
                    else if (which == 1)
                        Ko[((size_t)bh * 2048 + l) * 64 + d] = (bf16)v;
                    else  // V stored pre-transposed: [B*Hn, D, L]
                        Vo[((size_t)bh * 64 + d) * 2048 + l] = (bf16)v;
                }
            }
        }
    }
}

// ---------------------------------------------------------------------------
// Flash attention: 128 threads (4 waves), 64 query rows per block.
// Q/K bf16 [B*Hn, L=2048, D=64]; V bf16 [B*Hn, D, L]; O bf16 [B, L, Hn, D].
// Key blocks of 128, double-buffered async K/V tiles.
// ---------------------------------------------------------------------------
#define LDK 72    // K-tile stride (bf16): 144B, 16B-aligned, conflict-free
#define LDV 136   // V^T stride    (bf16): 272B, 16B-aligned, conflict-free
#define LDP 136   // P buffer stride (bf16): 272B, 16B-aligned, conflict-free

__global__ __launch_bounds__(128)
void attention(const bf16* __restrict__ Q, const bf16* __restrict__ Kg,
               const bf16* __restrict__ Vg, bf16* __restrict__ O)
{
    __shared__ bf16 kt[2][128 * LDK];
    __shared__ bf16 vt[2][64  * LDV];
    __shared__ bf16 pb[4 * 16 * LDP];

    const int tid  = threadIdx.x;
    const int lane = tid & 31;
    const int wave = tid >> 5;
    const int half = lane >> 4;
    const int l16  = lane & 15;
    const int bh   = blockIdx.y;                   // b*16 + h
    const int q0   = blockIdx.x * 64 + wave * 16;
    const size_t base = (size_t)bh * 2048 * 64;    // same for [L,D] and [D,L]

    auto stage = [&](int buf, int kb) {
        // K tile [128 keys][64 d]: 1024 x 16B chunks
#pragma unroll
        for (int t = 0; t < 8; ++t) {
            const int ch  = tid + t * 128;          // 0..1023
            const int key = ch >> 3;
            const int c   = (ch & 7) * 8;
            async_cp16(&Kg[base + (size_t)(kb + key) * 64 + c],
                       &kt[buf][key * LDK + c]);
        }
        // V^T tile [64 d][128 keys]: contiguous rows from [D,L] layout
#pragma unroll
        for (int t = 0; t < 8; ++t) {
            const int ch = tid + t * 128;           // 0..1023
            const int d  = ch >> 4;
            const int c  = (ch & 15) * 8;
            async_cp16(&Vg[base + (size_t)d * 2048 + kb + c],
                       &vt[buf][d * LDV + c]);
        }
    };

    // Q fragments (A layout), D=64 -> 2 chunks of K=32
    v16bf qf[2];
    {
        const bf16* qrow = Q + base + (size_t)(q0 + l16) * 64;
#pragma unroll
        for (int c = 0; c < 2; ++c) {
            const int off = c * 32 + half * 8;
            ((uint4*)&qf[c])[0] = *(const uint4*)&qrow[off];
            ((uint4*)&qf[c])[1] = *(const uint4*)&qrow[off + 16];
        }
    }

    float mrow[8], lrow[8];
#pragma unroll
    for (int r = 0; r < 8; ++r) { mrow[r] = -1e30f; lrow[r] = 0.f; }
    v8f oacc[4];
#pragma unroll
    for (int jj = 0; jj < 4; ++jj)
#pragma unroll
        for (int r = 0; r < 8; ++r) oacc[jj][r] = 0.f;

    bf16* myP = &pb[wave * 16 * LDP];

    stage(0, 0);
    async_wait0();
    __syncthreads();

    for (int ib = 0; ib < 16; ++ib) {
        const int cur = ib & 1;
        // Prefetch next key block; buffer cur^1 was last read before the
        // previous iteration's end barrier, so overwrite is safe.
        if (ib + 1 < 16) stage(cur ^ 1, (ib + 1) * 128);

        // --- S = Q @ K^T : 8 key n-tiles ---
        v8f s[8];
#pragma unroll
        for (int j = 0; j < 8; ++j) {
            v8f a;
#pragma unroll
            for (int r = 0; r < 8; ++r) a[r] = 0.f;
#pragma unroll
            for (int c = 0; c < 2; ++c) {
                v16bf kf;
                const int off = (j * 16 + l16) * LDK + c * 32 + half * 8;
                ((uint4*)&kf)[0] = *(const uint4*)&kt[cur][off];
                ((uint4*)&kf)[1] = *(const uint4*)&kt[cur][off + 16];
                a = __builtin_amdgcn_wmma_f32_16x16x32_bf16(
                        false, qf[c], false, kf, (short)0, a, false, false);
            }
            s[j] = a;
        }

        // --- online softmax (rows live in fixed VGPR slots) ---
        float alpha[8];
#pragma unroll
        for (int r = 0; r < 8; ++r) {
            float mx = s[0][r];
#pragma unroll
            for (int j = 1; j < 8; ++j) mx = fmaxf(mx, s[j][r]);
            mx = fmaxf(mx, __shfl_xor(mx, 1, 32));
            mx = fmaxf(mx, __shfl_xor(mx, 2, 32));
            mx = fmaxf(mx, __shfl_xor(mx, 4, 32));
            mx = fmaxf(mx, __shfl_xor(mx, 8, 32));
            const float mnew = fmaxf(mrow[r], mx);
            alpha[r] = exp2f(mrow[r] - mnew);
            mrow[r]  = mnew;
            float sum = 0.f;
#pragma unroll
            for (int j = 0; j < 8; ++j) {
                const float p = exp2f(s[j][r] - mnew);
                s[j][r] = p;
                sum += p;
            }
            sum += __shfl_xor(sum, 1, 32);
            sum += __shfl_xor(sum, 2, 32);
            sum += __shfl_xor(sum, 4, 32);
            sum += __shfl_xor(sum, 8, 32);
            lrow[r] = lrow[r] * alpha[r] + sum;
        }
#pragma unroll
        for (int jj = 0; jj < 4; ++jj)
#pragma unroll
            for (int r = 0; r < 8; ++r) oacc[jj][r] *= alpha[r];

        // --- P (C layout) -> LDS bf16, reload as A fragments ---
#pragma unroll
        for (int j = 0; j < 8; ++j)
#pragma unroll
            for (int r = 0; r < 8; ++r)
                myP[(half * 8 + r) * LDP + j * 16 + l16] = (bf16)s[j][r];
        __syncthreads();

        // --- O += P @ V : 4 d n-tiles, 4 key k-chunks ---
#pragma unroll
        for (int c = 0; c < 4; ++c) {
            v16bf pf;
            const int off = l16 * LDP + c * 32 + half * 8;
            ((uint4*)&pf)[0] = *(const uint4*)&myP[off];
            ((uint4*)&pf)[1] = *(const uint4*)&myP[off + 16];
#pragma unroll
            for (int jj = 0; jj < 4; ++jj) {
                v16bf vf;
                const int voff = (jj * 16 + l16) * LDV + c * 32 + half * 8;
                ((uint4*)&vf)[0] = *(const uint4*)&vt[cur][voff];
                ((uint4*)&vf)[1] = *(const uint4*)&vt[cur][voff + 16];
                oacc[jj] = __builtin_amdgcn_wmma_f32_16x16x32_bf16(
                               false, pf, false, vf, (short)0, oacc[jj], false, false);
            }
        }

        async_wait0();          // next K/V tiles landed
        __syncthreads();        // all waves done with cur; next buffer visible
    }

    // --- normalize and store O as [B, L, Hn, D] bf16 ---
    const int bb = bh >> 4, h = bh & 15;
#pragma unroll
    for (int jj = 0; jj < 4; ++jj) {
        const int d = jj * 16 + l16;
#pragma unroll
        for (int r = 0; r < 8; ++r) {
            const int gq = q0 + half * 8 + r;
            const float v = oacc[jj][r] / lrow[r];
            O[(((size_t)(bb * 2048 + gq)) * 16 + h) * 64 + d] = (bf16)v;
        }
    }
}

// ---------------------------------------------------------------------------
extern "C" void kernel_launch(void* const* d_in, const int* in_sizes, int n_in,
                              void* d_out, int out_size, void* d_ws, size_t ws_size,
                              hipStream_t stream)
{
    const float* x      = (const float*)d_in[0];
    const float* qkv_w  = (const float*)d_in[1];
    const float* qkv_b  = (const float*)d_in[2];
    const float* proj_w = (const float*)d_in[3];
    const float* proj_b = (const float*)d_in[4];
    const float* fc1_w  = (const float*)d_in[5];
    const float* fc1_b  = (const float*)d_in[6];
    const float* fc2_w  = (const float*)d_in[7];
    const float* fc2_b  = (const float*)d_in[8];
    float* out = (float*)d_out;

    constexpr int M = 8192;        // B*L
    constexpr int C = 1024;
    constexpr int N3 = 3072;
    constexpr int HH = 4096;

    char* p = (char*)d_ws;
    auto alloc = [&](size_t bytes) -> char* {
        char* r = p;
        p += (bytes + 255) & ~(size_t)255;
        return r;
    };
    bf16* Wqkv = (bf16*)alloc((size_t)N3 * C * 2);
    bf16* Wprj = (bf16*)alloc((size_t)C  * C * 2);
    bf16* Wfc1 = (bf16*)alloc((size_t)HH * C * 2);
    bf16* Wfc2 = (bf16*)alloc((size_t)C  * HH * 2);
    bf16* hbuf = (bf16*)alloc((size_t)M * C * 2);
    bf16* Qb   = (bf16*)alloc((size_t)M * C * 2);
    bf16* Kb   = (bf16*)alloc((size_t)M * C * 2);
    bf16* Vb   = (bf16*)alloc((size_t)M * C * 2);
    bf16* Ob   = (bf16*)alloc((size_t)M * C * 2);
    float* x1  = (float*)alloc((size_t)M * C * 4);
    bf16* h2   = (bf16*)alloc((size_t)M * C * 2);
    bf16* a1   = (bf16*)alloc((size_t)M * HH * 2);

    // 1) weights -> bf16, transposed to [N][K]
    transpose_to_bf16<<<dim3(N3 / 32, C / 32), 256, 0, stream>>>(qkv_w, Wqkv, C, N3);
    transpose_to_bf16<<<dim3(C  / 32, C / 32), 256, 0, stream>>>(proj_w, Wprj, C, C);
    transpose_to_bf16<<<dim3(HH / 32, C / 32), 256, 0, stream>>>(fc1_w, Wfc1, C, HH);
    transpose_to_bf16<<<dim3(C  / 32, HH / 32), 256, 0, stream>>>(fc2_w, Wfc2, HH, C);

    // 2) LN1
    layernorm_bf16<<<M, 256, 0, stream>>>(x, hbuf);

    // 3) QKV GEMM + scatter (Q pre-scaled; V stored [B*Hn, D, L])
    gemm_bf16<EP_QKV><<<dim3(N3 / 128, M / 128), 256, 0, stream>>>(
        hbuf, Wqkv, qkv_b, nullptr, nullptr, nullptr, Qb, Kb, Vb, M, N3, C);

    // 4) attention
    attention<<<dim3(2048 / 64, 4 * 16), 128, 0, stream>>>(Qb, Kb, Vb, Ob);

    // 5) proj GEMM + residual -> x1 (f32)
    gemm_bf16<EP_RESID><<<dim3(C / 128, M / 128), 256, 0, stream>>>(
        Ob, Wprj, proj_b, x, x1, nullptr, nullptr, nullptr, nullptr, M, C, C);

    // 6) LN2
    layernorm_bf16<<<M, 256, 0, stream>>>(x1, h2);

    // 7) fc1 GEMM + gelu -> a1 (bf16)
    gemm_bf16<EP_GELU><<<dim3(HH / 128, M / 128), 256, 0, stream>>>(
        h2, Wfc1, fc1_b, nullptr, nullptr, a1, nullptr, nullptr, nullptr, M, HH, C);

    // 8) fc2 GEMM + residual -> out (f32)
    gemm_bf16<EP_RESID><<<dim3(C / 128, M / 128), 256, 0, stream>>>(
        a1, Wfc2, fc2_b, x1, out, nullptr, nullptr, nullptr, nullptr, M, C, HH);
}
